// GGE_11957188952712
// MI455X (gfx1250) — matrix operand order repaired and checked by reading, not compiled, not run on hardware
//
#include <hip/hip_runtime.h>

// ---------------------------------------------------------------------------
// CDNA5 / gfx1250 implementation. wave32, WMMA bf16 16x16x32, fp32 accumulate.
// B-tiles optionally moved by the Tensor Data Mover (tensor_load_to_lds).
// ---------------------------------------------------------------------------

typedef __attribute__((ext_vector_type(16))) __bf16       v16bf;
typedef __attribute__((ext_vector_type(8)))  __bf16       v8bf;
typedef __attribute__((ext_vector_type(8)))  float        v8f;
typedef __attribute__((ext_vector_type(4)))  unsigned int v4u;
typedef __attribute__((ext_vector_type(8)))  unsigned int v8u;
typedef __attribute__((ext_vector_type(4)))  int          v4i;
typedef __attribute__((ext_vector_type(8)))  int          v8i;

#define BB    4
#define NPTS  2048
#define CC    256
#define KNN   20
#define RAD2  0.25f
#define EPSV  1e-5f

#if defined(__has_builtin)
#  if __has_builtin(__builtin_amdgcn_tensor_load_to_lds) && \
      __has_builtin(__builtin_amdgcn_s_wait_tensorcnt)
#    define GGE_HAVE_TDM 1
#  endif
#endif
#ifndef GGE_HAVE_TDM
#  define GGE_HAVE_TDM 0
#endif

static __device__ __forceinline__ float  bf2f(__bf16 x) { return (float)x; }
static __device__ __forceinline__ __bf16 f2bf(float x)  { return (__bf16)x; }

// ---- WMMA fragment loaders (contiguous k order, same for A and B, wave32) --------------
// A (16x32 bf16), row stride 64 in LDS: lane row = lane&15; lanes 0-15 carry
// K 0..15, lanes 16-31 carry K 16..31  -> two 16B LDS loads.
static __device__ __forceinline__ v16bf frag_a(const __bf16* p, int lane) {
    const int row = lane & 15;
    const int h8  = (lane >= 16) ? 16 : 0;
    const __bf16* q = p + row * 64 + h8;
    v4u a = *(const v4u*)q;          // K  0.. 7 (lo lanes) / 16..23 (hi lanes)
    v4u b = *(const v4u*)(q + 8);    // K  8..15 (lo lanes) / 24..31 (hi lanes)
    v8u u = __builtin_shufflevector(a, b, 0, 1, 2, 3, 4, 5, 6, 7);
    return __builtin_bit_cast(v16bf, u);
}
// B (32x16 bf16): lanes 0-15 K=0..15 of column lane&15, lanes 16-31 K=16..31
// -> one 32B contiguous LDS load.
static __device__ __forceinline__ v16bf frag_b(const __bf16* p, int lane) {
    const int col = lane & 15;
    const int kb  = (lane >= 16) ? 16 : 0;
    v8u u = *(const v8u*)(p + col * 64 + kb);
    return __builtin_bit_cast(v16bf, u);
}

// ---- Tensor Data Mover: 2-D 64x64 bf16 tile, row stride CA elements --------
#if GGE_HAVE_TDM
static __device__ __forceinline__ void tdm_load_b_tile(unsigned lds_off,
                                                       const __bf16* gptr, int CA) {
    const unsigned long long ga = (unsigned long long)(__SIZE_TYPE__)gptr;
    // D# group 0 (ISA 8.3): count=1 | lds_addr | global_addr[56:0] | type=2
    v4u g0;
    g0[0] = 1u;
    g0[1] = lds_off;
    g0[2] = (unsigned)(ga & 0xffffffffu);
    g0[3] = (unsigned)((ga >> 32) & 0x01ffffffu) | (2u << 30);
    // D# group 1 (ISA 8.4): data_size=1(2B); tensor 64x64 (tile-exact, always
    // in-bounds); tile 64x64; tensor_dim0_stride = CA.
    const unsigned td0 = 64u, td1 = 64u, tl0 = 64u, tl1 = 64u;
    const unsigned st0 = (unsigned)CA;
    v8i g1;
    g1[0] = (int)(1u << 16);                                   // data_size=2B
    g1[1] = (int)((td0 & 0xffffu) << 16);                      // tensor_dim0 lo
    g1[2] = (int)(((td0 >> 16) & 0xffffu) | ((td1 & 0xffffu) << 16));
    g1[3] = (int)(((td1 >> 16) & 0xffffu) | ((tl0 & 0xffffu) << 16));
    g1[4] = (int)(tl1 & 0xffffu);                              // tile_dim1, dim2=0
    g1[5] = (int)st0;                                          // stride0 lo
    g1[6] = 0;                                                 // stride0 hi, stride1 lo
    g1[7] = 0;
    v4i g2 = {0, 0, 0, 0}, g3 = {0, 0, 0, 0};                  // 2-D: unused
#if __clang_major__ >= 23
    v8i g4 = {0, 0, 0, 0, 0, 0, 0, 0};
    __builtin_amdgcn_tensor_load_to_lds(g0, g1, g2, g3, g4, 0);
#else
    __builtin_amdgcn_tensor_load_to_lds(g0, g1, g2, g3, 0);
#endif
}
#endif

// ---------------------------------------------------------------------------
// Kernel: kNN (top-21, drop self) + PPF (first 64 in-radius indices ascending,
// max angle).  One thread per query point; coords tiled through LDS.
// ---------------------------------------------------------------------------
__global__ __launch_bounds__(256)
void gge_knn_ppf(const float* __restrict__ coords, const float* __restrict__ normals,
                 int* __restrict__ knn, float* __restrict__ ppf) {
    const int b = blockIdx.y;
    const int n = blockIdx.x * blockDim.x + threadIdx.x;
    const float* cb = coords  + (size_t)b * 3 * NPTS;
    const float* nb = normals + (size_t)b * 3 * NPTS;

    const float px = cb[n], py = cb[NPTS + n], pz = cb[2 * NPTS + n];
    const float nx = nb[n], ny = nb[NPTS + n], nz = nb[2 * NPTS + n];

    float bd[KNN + 1];
    int   bi[KNN + 1];
#pragma unroll
    for (int i = 0; i <= KNN; ++i) { bd[i] = 3.402823e38f; bi[i] = 0; }

    int   rcount = 0;
    float maxang = -1e30f;

    __shared__ float sx[256], sy[256], sz[256];
    for (int m0 = 0; m0 < NPTS; m0 += 256) {
        __syncthreads();
        const int mi = m0 + threadIdx.x;
        sx[threadIdx.x] = cb[mi];
        sy[threadIdx.x] = cb[NPTS + mi];
        sz[threadIdx.x] = cb[2 * NPTS + mi];
        __syncthreads();
        for (int j = 0; j < 256; ++j) {
            const float dx = sx[j] - px, dy = sy[j] - py, dz = sz[j] - pz;
            const float d  = dx * dx + dy * dy + dz * dz;
            const int   m  = m0 + j;
            if (d < bd[KNN]) {                       // stable top-(k+1) insert
                int p = KNN;
                while (p > 0 && bd[p - 1] > d) { bd[p] = bd[p - 1]; bi[p] = bi[p - 1]; --p; }
                bd[p] = d; bi[p] = m;
            }
            if (d <= RAD2 && rcount < 64) {          // first 64 in-radius (asc m)
                ++rcount;
                const float cxv = ny * dz - nz * dy;
                const float cyv = nz * dx - nx * dz;
                const float czv = nx * dy - ny * dx;
                const float cn  = sqrtf(cxv * cxv + cyv * cyv + czv * czv + 1e-12f);
                maxang = fmaxf(maxang, atan2f(cn, nx * dx + ny * dy + nz * dz));
            }
        }
    }
#pragma unroll
    for (int k = 0; k < KNN; ++k)
        knn[((size_t)b * NPTS + n) * KNN + k] = bi[k + 1];   // drop self
    ppf[(size_t)b * NPTS + n] = maxang;
}

// ---------------------------------------------------------------------------
// WMMA GEMM:  Y[b, r, o] = sum_c A'[b,r,c] * W[o,c]     (bf16 in, f32 acc)
//   EDGE : A' = [ feat[n] || feat[nbr]-feat[n] ] built on the fly; r=n*KNN+k.
//   Block = 128 threads (4 waves): 16-row x 64-col tile, K-step 64.
//   The c<Cin branch is UNIFORM per K-step (kc,Cin multiples of 64), so A
//   staging is divergence-free 16B vector traffic. B tile comes from the TDM
//   when available, else a cooperative vector copy. IN statistics fused.
// ---------------------------------------------------------------------------
template <bool EDGE, bool TDM>
__global__ __launch_bounds__(128)
void gge_gemm(const __bf16* __restrict__ A, const int* __restrict__ nbr,
              const __bf16* __restrict__ W, __bf16* __restrict__ Y,
              float* __restrict__ ssum, float* __restrict__ ssq,
              int R, int CA, int Cin, int Cout) {
    __shared__ __bf16 lA[16 * 64];
    __shared__ __bf16 lB[64 * 64];

    const int b    = blockIdx.z;
    const int r0   = blockIdx.x * 16;
    const int o0   = blockIdx.y * 64;
    const int tid  = threadIdx.x;
    const int lane = tid & 31;
    const int wave = tid >> 5;
    const bool hi  = lane >= 16;

    // Per-thread A-tile ownership: row = tid>>3, 8 contiguous bf16 at aseg.
    const int arow = tid >> 3;
    const int aseg = (tid & 7) * 8;

    const __bf16* actr   = A;   // EDGE: center row base
    const __bf16* anbrp  = A;   // EDGE: neighbor row base
    const __bf16* aplain = A + ((size_t)b * R + r0 + arow) * CA;
    if (EDGE) {
        const int gr = r0 + arow;
        const int n  = gr / KNN;
        const int nn = nbr[((size_t)b * NPTS + n) * KNN + (gr - n * KNN)];
        actr  = A + ((size_t)b * NPTS + n)  * Cin;
        anbrp = A + ((size_t)b * NPTS + nn) * Cin;
    }

    v8f acc = {};

    for (int kc = 0; kc < CA; kc += 64) {
#if GGE_HAVE_TDM
        if constexpr (TDM) {
            if (wave == 0)
                tdm_load_b_tile((unsigned)(__SIZE_TYPE__)(__bf16*)lB,
                                W + (size_t)o0 * CA + kc, CA);
        }
#endif
        // ---- stage A tile (16 x 64): 16B per thread, uniform branch -------
        if (EDGE) {
            if (kc < Cin) {
                *(v8bf*)(lA + arow * 64 + aseg) = *(const v8bf*)(actr + kc + aseg);
            } else {
                const int cc = kc - Cin + aseg;
                v8bf uc = *(const v8bf*)(actr + cc);
                v8bf un = *(const v8bf*)(anbrp + cc);
                v8bf d;
#pragma unroll
                for (int w = 0; w < 8; ++w) d[w] = f2bf(bf2f(un[w]) - bf2f(uc[w]));
                *(v8bf*)(lA + arow * 64 + aseg) = d;
            }
        } else {
            *(v8bf*)(lA + arow * 64 + aseg) = *(const v8bf*)(aplain + kc + aseg);
            if (kc + 64 < CA) __builtin_prefetch(aplain + kc + 64 + aseg, 0, 1);
        }
        // ---- stage B tile (64 x 64) when no TDM ---------------------------
        if constexpr (!TDM) {
#pragma unroll
            for (int i = 0; i < 4; ++i) {
                const int idx8 = i * 128 + tid;            // [0,512) 8-elt spans
                const int ocol = idx8 >> 3;
                const int cseg = (idx8 & 7) * 8;
                *(v8bf*)(lB + ocol * 64 + cseg) =
                    *(const v8bf*)(W + (size_t)(o0 + ocol) * CA + kc + cseg);
            }
        }
#if GGE_HAVE_TDM
        if constexpr (TDM) {
            if (wave == 0) __builtin_amdgcn_s_wait_tensorcnt(0);
        }
#endif
        __syncthreads();

        const __bf16* lb = lB + wave * (16 * 64);
        v16bf a0 = frag_a(lA,      lane);
        v16bf b0 = frag_b(lb,      lane);
        acc = __builtin_amdgcn_wmma_f32_16x16x32_bf16(false, a0, false, b0,
                                                      (short)0, acc, false, false);
        v16bf a1 = frag_a(lA + 32, lane);
        v16bf b1 = frag_b(lb + 32, lane);
        acc = __builtin_amdgcn_wmma_f32_16x16x32_bf16(false, a1, false, b1,
                                                      (short)0, acc, false, false);
        __syncthreads();
    }

    // ---- epilogue: store bf16 tile + fused instance-norm statistics -------
    const int col = o0 + wave * 16 + (lane & 15);  // C/D: N = lane&15, M = j(+8)
    float s = 0.f, sq = 0.f;
#pragma unroll
    for (int j = 0; j < 8; ++j) {
        const int M = j + (hi ? 8 : 0);
        const float v = acc[j];
        Y[((size_t)b * R + r0 + M) * Cout + col] = f2bf(v);
        s += v; sq += v * v;
    }
    s  += __shfl_xor(s, 16, 32);                   // fold lanes 16-31 (wave32)
    sq += __shfl_xor(sq, 16, 32);
    if (!hi) {
        atomicAdd(&ssum[b * Cout + col], s);
        atomicAdd(&ssq [b * Cout + col], sq);
    }
}

// ---------------------------------------------------------------------------
// Instance-norm over (n,k) + leaky-ReLU + max over k  ->  (B,N,Cout) bf16
// ---------------------------------------------------------------------------
__global__ __launch_bounds__(256)
void gge_norm_max(const __bf16* __restrict__ Y, const float* __restrict__ ssum,
                  const float* __restrict__ ssq, __bf16* __restrict__ dstT, int Cout) {
    const size_t t = (size_t)blockIdx.x * blockDim.x + threadIdx.x;
    if (t >= (size_t)BB * NPTS * Cout) return;
    const int col = (int)(t % Cout);
    const int n   = (int)((t / Cout) % NPTS);
    const int b   = (int)(t / ((size_t)Cout * NPTS));
    const float cnt = (float)(NPTS * KNN);
    const float m   = ssum[b * Cout + col] / cnt;
    const float var = ssq [b * Cout + col] / cnt - m * m;
    const float rs  = rsqrtf(fmaxf(var, 0.f) + EPSV);
    float mx = -3.4e38f;
    const __bf16* yb = Y + ((size_t)b * NPTS * KNN + (size_t)n * KNN) * Cout + col;
#pragma unroll
    for (int k = 0; k < KNN; ++k) {
        float v = (bf2f(yb[(size_t)k * Cout]) - m) * rs;
        v = (v >= 0.f) ? v : 0.2f * v;
        mx = fmaxf(mx, v);
    }
    dstT[t] = f2bf(mx);
}

// ---------------------------------------------------------------------------
// Instance-norm over n + leaky-ReLU. Optional bf16 store (ld/offset for concat
// targets) and optional transposed fp32 store (final output).
// Biases are omitted: IN subtracts the mean, so per-channel biases cancel.
// ---------------------------------------------------------------------------
__global__ __launch_bounds__(256)
void gge_norm_row(const __bf16* __restrict__ Y, const float* __restrict__ ssum,
                  const float* __restrict__ ssq, int Cout,
                  __bf16* __restrict__ dst, int dld, int doff,
                  float* __restrict__ outT) {
    const size_t t = (size_t)blockIdx.x * blockDim.x + threadIdx.x;
    if (t >= (size_t)BB * NPTS * Cout) return;
    const int col = (int)(t % Cout);
    const int n   = (int)((t / Cout) % NPTS);
    const int b   = (int)(t / ((size_t)Cout * NPTS));
    const float cnt = (float)NPTS;
    const float m   = ssum[b * Cout + col] / cnt;
    const float var = ssq [b * Cout + col] / cnt - m * m;
    const float rs  = rsqrtf(fmaxf(var, 0.f) + EPSV);
    float v = (bf2f(Y[((size_t)b * NPTS + n) * Cout + col]) - m) * rs;
    v = (v >= 0.f) ? v : 0.2f * v;
    if (dst)  dst[((size_t)b * NPTS + n) * dld + doff + col] = f2bf(v);
    if (outT) outT[((size_t)b * Cout + col) * NPTS + n] = v;
}

// ---------------------------------------------------------------------------
// Small utility kernels
// ---------------------------------------------------------------------------
__global__ __launch_bounds__(256)
void gge_cvt_w(const float* __restrict__ src, __bf16* __restrict__ dst,
               int rows, int scols, int dld) {
    const size_t t = (size_t)blockIdx.x * blockDim.x + threadIdx.x;
    if (t >= (size_t)rows * dld) return;
    const int c = (int)(t % dld);
    const int r = (int)(t / dld);
    dst[t] = f2bf((c < scols) ? src[(size_t)r * scols + c] : 0.f);
}

__global__ __launch_bounds__(256)
void gge_transpose_feats(const float* __restrict__ feats, __bf16* __restrict__ featsT) {
    const size_t t = (size_t)blockIdx.x * blockDim.x + threadIdx.x;
    if (t >= (size_t)BB * NPTS * CC) return;
    const int c = (int)(t % CC);
    const int n = (int)((t / CC) % NPTS);
    const int b = (int)(t / ((size_t)CC * NPTS));
    featsT[t] = f2bf(feats[((size_t)b * CC + c) * NPTS + n]);
}

__global__ __launch_bounds__(256)
void gge_f3cat(const __bf16* __restrict__ featsT, const __bf16* __restrict__ f1T,
               const __bf16* __restrict__ f2T, __bf16* __restrict__ f3T) {
    const size_t t = (size_t)blockIdx.x * blockDim.x + threadIdx.x;
    if (t >= (size_t)BB * NPTS * 1024) return;
    const int c = (int)(t % 1024);
    const size_t bn = t / 1024;
    __bf16 v;
    if      (c < 256) v = featsT[bn * 256 + c];
    else if (c < 512) v = f1T[bn * 256 + (c - 256)];
    else              v = f2T[bn * 512 + (c - 512)];
    f3T[t] = v;
}

__global__ __launch_bounds__(256)
void gge_ppf_h0(const float* __restrict__ ppf, __bf16* __restrict__ h0) {
    const size_t t = (size_t)blockIdx.x * blockDim.x + threadIdx.x;
    if (t >= (size_t)BB * NPTS) return;
    h0[t * 320] = f2bf(ppf[t]);   // column 0 of padded h0 (ld = 320)
}

// ---------------------------------------------------------------------------
// Host side
// ---------------------------------------------------------------------------
extern "C" void kernel_launch(void* const* d_in, const int* in_sizes, int n_in,
                              void* d_out, int out_size, void* d_ws, size_t ws_size,
                              hipStream_t stream) {
    (void)in_sizes; (void)n_in; (void)out_size;
    const float* coords  = (const float*)d_in[0];
    const float* feats   = (const float*)d_in[1];
    const float* normals = (const float*)d_in[2];
    const float* w1      = (const float*)d_in[3];
    const float* w2      = (const float*)d_in[4];
    const float* w3      = (const float*)d_in[5];
    const float* fw1     = (const float*)d_in[6];   // fused_b1/b2 cancel under IN
    const float* fw2     = (const float*)d_in[8];
    float* out = (float*)d_out;

    char* ws = (char*)d_ws;
    size_t off = 0;
    auto take = [&](size_t bytes) { size_t o = off; off = (off + bytes + 255) & ~(size_t)255; return o; };

    const size_t oFeatsT = take((size_t)BB * NPTS * 256 * 2);
    const size_t oF1T    = take((size_t)BB * NPTS * 256 * 2);
    const size_t oF2T    = take((size_t)BB * NPTS * 512 * 2);
    const size_t oF3T    = take((size_t)BB * NPTS * 1024 * 2);
    const size_t oH0     = take((size_t)BB * NPTS * 320 * 2);
    const size_t oH1     = take((size_t)BB * NPTS * 512 * 2);
    const size_t oKnn    = take((size_t)BB * NPTS * KNN * 4);
    const size_t oPpf    = take((size_t)BB * NPTS * 4);
    const size_t oW1b    = take((size_t)256 * 512 * 2);
    const size_t oW2b    = take((size_t)512 * 512 * 2);
    const size_t oW3b    = take((size_t)256 * 1024 * 2);
    const size_t oFw1b   = take((size_t)512 * 320 * 2);
    const size_t oFw2b   = take((size_t)256 * 512 * 2);
    const size_t oSum    = take((size_t)BB * 512 * 4);
    const size_t oSq     = take((size_t)BB * 512 * 4);
    const size_t oY      = take((size_t)BB * NPTS * KNN * 512 * 2);   // biggest y
    if (ws_size < off) return;

    __bf16* featsT = (__bf16*)(ws + oFeatsT);
    __bf16* f1T    = (__bf16*)(ws + oF1T);
    __bf16* f2T    = (__bf16*)(ws + oF2T);
    __bf16* f3T    = (__bf16*)(ws + oF3T);
    __bf16* h0     = (__bf16*)(ws + oH0);
    __bf16* h1     = (__bf16*)(ws + oH1);
    int*    knn    = (int*)   (ws + oKnn);
    float*  ppf    = (float*) (ws + oPpf);
    __bf16* w1b    = (__bf16*)(ws + oW1b);
    __bf16* w2b    = (__bf16*)(ws + oW2b);
    __bf16* w3b    = (__bf16*)(ws + oW3b);
    __bf16* fw1b   = (__bf16*)(ws + oFw1b);
    __bf16* fw2b   = (__bf16*)(ws + oFw2b);
    float*  ssum   = (float*) (ws + oSum);
    float*  ssq    = (float*) (ws + oSq);
    __bf16* ybuf   = (__bf16*)(ws + oY);

    const int T = 256;
    auto blk = [](size_t total, int tpb) { return (int)((total + tpb - 1) / tpb); };
    auto zstats = [&]() {
        hipMemsetAsync(ssum, 0, (size_t)BB * 512 * 4, stream);
        hipMemsetAsync(ssq,  0, (size_t)BB * 512 * 4, stream);
    };
    constexpr bool TDM = (GGE_HAVE_TDM != 0);

    // weights -> bf16 (fused_w1 padded 257 -> 320 so CA % 64 == 0)
    gge_cvt_w<<<blk((size_t)256 * 512, T), T, 0, stream>>>(w1,  w1b,  256, 512, 512);
    gge_cvt_w<<<blk((size_t)512 * 512, T), T, 0, stream>>>(w2,  w2b,  512, 512, 512);
    gge_cvt_w<<<blk((size_t)256 * 1024, T), T, 0, stream>>>(w3,  w3b,  256, 1024, 1024);
    gge_cvt_w<<<blk((size_t)512 * 320, T), T, 0, stream>>>(fw1, fw1b, 512, 257, 320);
    gge_cvt_w<<<blk((size_t)256 * 512, T), T, 0, stream>>>(fw2, fw2b, 256, 512, 512);

    gge_transpose_feats<<<blk((size_t)BB * NPTS * CC, T), T, 0, stream>>>(feats, featsT);
    gge_knn_ppf<<<dim3(NPTS / 256, BB), 256, 0, stream>>>(coords, normals, knn, ppf);

    const int REdge = NPTS * KNN;                                   // 40960
    // --- GCN layer 1: (B,N,K,512) x (256,512)^T -------------------------
    zstats();
    gge_gemm<true, TDM><<<dim3(REdge / 16, 256 / 64, BB), 128, 0, stream>>>(
        featsT, knn, w1b, ybuf, ssum, ssq, REdge, 512, 256, 256);
    gge_norm_max<<<blk((size_t)BB * NPTS * 256, T), T, 0, stream>>>(ybuf, ssum, ssq, f1T, 256);

    // --- GCN layer 2: (B,N,K,512) x (512,512)^T -------------------------
    zstats();
    gge_gemm<true, TDM><<<dim3(REdge / 16, 512 / 64, BB), 128, 0, stream>>>(
        f1T, knn, w2b, ybuf, ssum, ssq, REdge, 512, 256, 512);
    gge_norm_max<<<blk((size_t)BB * NPTS * 512, T), T, 0, stream>>>(ybuf, ssum, ssq, f2T, 512);

    // --- f3 = [feats | f1 | f2] ; y3 = f3 x w3^T ------------------------
    gge_f3cat<<<blk((size_t)BB * NPTS * 1024, T), T, 0, stream>>>(featsT, f1T, f2T, f3T);
    zstats();
    gge_gemm<false, TDM><<<dim3(NPTS / 16, 256 / 64, BB), 128, 0, stream>>>(
        f3T, nullptr, w3b, ybuf, ssum, ssq, NPTS, 1024, 0, 256);
    hipMemsetAsync(h0, 0, (size_t)BB * NPTS * 320 * 2, stream);
    gge_ppf_h0<<<blk((size_t)BB * NPTS, T), T, 0, stream>>>(ppf, h0);
    gge_norm_row<<<blk((size_t)BB * NPTS * 256, T), T, 0, stream>>>(
        ybuf, ssum, ssq, 256, h0, 320, 1, nullptr);                 // gcn -> h0[:,1:257]

    // --- fused layer 1: h0 (B,N,320 padded) x fw1 (512,320) -------------
    zstats();
    gge_gemm<false, TDM><<<dim3(NPTS / 16, 512 / 64, BB), 128, 0, stream>>>(
        h0, nullptr, fw1b, ybuf, ssum, ssq, NPTS, 320, 0, 512);
    gge_norm_row<<<blk((size_t)BB * NPTS * 512, T), T, 0, stream>>>(
        ybuf, ssum, ssq, 512, h1, 512, 0, nullptr);

    // --- fused layer 2: h1 x fw2 (256,512) -> transposed fp32 output ----
    zstats();
    gge_gemm<false, TDM><<<dim3(NPTS / 16, 256 / 64, BB), 128, 0, stream>>>(
        h1, nullptr, fw2b, ybuf, ssum, ssq, NPTS, 512, 0, 256);
    gge_norm_row<<<blk((size_t)BB * NPTS * 256, T), T, 0, stream>>>(
        ybuf, ssum, ssq, 256, nullptr, 0, 0, out);
}
